// AttentionBlock_51685636440106
// MI455X (gfx1250) — matrix operand order repaired and checked by reading, not compiled, not run on hardware
//
#include <hip/hip_runtime.h>
#include <hip/hip_bf16.h>

// ---------------------------------------------------------------------------
// Types for CDNA5 WMMA (wave32)
// ---------------------------------------------------------------------------
typedef __attribute__((ext_vector_type(16))) _Float16 v16h;
typedef __attribute__((ext_vector_type(8)))  _Float16 v8h;
typedef __attribute__((ext_vector_type(8)))  float    v8f;
typedef __attribute__((ext_vector_type(4)))  int      v4i;

union F16x16 {
    v16h v;
    v8h  h8[2];
};

// Async global->LDS path (gfx1250). Guarded so the file also compiles on
// toolchains that do not declare these builtins (falls back to
// global_load_b128 + ds_store_b128, same LDS tiling).
#if __has_builtin(__builtin_amdgcn_global_load_async_to_lds_b128) && \
    __has_builtin(__builtin_amdgcn_s_wait_asynccnt)
#define HAS_ASYNC_LDS 1
#else
#define HAS_ASYNC_LDS 0
#endif

#if HAS_ASYNC_LDS
typedef __attribute__((address_space(1))) v4i gv4i;   // global int4
typedef __attribute__((address_space(3))) v4i sv4i;   // LDS int4
#endif

__device__ __forceinline__ void copy16_g2lds(const _Float16* gsrc,
                                             _Float16* ldst) {
#if HAS_ASYNC_LDS
    __builtin_amdgcn_global_load_async_to_lds_b128(
        (gv4i*)gsrc, (sv4i*)ldst, 0, 0);
#else
    *(v8h*)ldst = *(const v8h*)gsrc;
#endif
}

#if HAS_ASYNC_LDS
#define WAIT_ASYNC(N) __builtin_amdgcn_s_wait_asynccnt(N)
#else
#define WAIT_ASYNC(N) ((void)0)
#endif

// ---------------------------------------------------------------------------
// Problem constants (B, S, H from the reference)
// ---------------------------------------------------------------------------
namespace cfg {
constexpr int Bn = 4;
constexpr int S  = 2048;
constexpr int H  = 1024;
constexpr int F  = 4 * H;       // 4096
constexpr int M  = Bn * S;      // 8192 rows for all GEMMs
constexpr float LN_EPS = 1e-5f;
}

// ---------------------------------------------------------------------------
// f32 -> f16 convert (grid-stride)
// ---------------------------------------------------------------------------
__global__ void cvt_f32_to_f16_kernel(const float* __restrict__ src,
                                      _Float16* __restrict__ dst, size_t n) {
    size_t i = (size_t)blockIdx.x * blockDim.x + threadIdx.x;
    size_t stride = (size_t)gridDim.x * blockDim.x;
    for (; i < n; i += stride) dst[i] = (_Float16)src[i];
}

// ---------------------------------------------------------------------------
// Transpose + convert: W[K][N] f32  ->  WT[N][K] f16   (32x32 LDS tile)
// block: (32, 8); grid: (N/32, K/32)
// ---------------------------------------------------------------------------
__global__ void transpose_cvt_kernel(const float* __restrict__ W,
                                     _Float16* __restrict__ WT,
                                     int Krows, int Ncols) {
    __shared__ float tile[32][33];
    const int bn = blockIdx.x * 32;   // N base
    const int bk = blockIdx.y * 32;   // K base
    const int tx = threadIdx.x;       // 0..31
    const int ty = threadIdx.y;       // 0..7
#pragma unroll
    for (int i = 0; i < 32; i += 8)
        tile[ty + i][tx] = W[(size_t)(bk + ty + i) * Ncols + (bn + tx)];
    __syncthreads();
#pragma unroll
    for (int i = 0; i < 32; i += 8)
        WT[(size_t)(bn + ty + i) * Krows + (bk + tx)] = (_Float16)tile[tx][ty + i];
}

// ---------------------------------------------------------------------------
// WMMA GEMM:  C[M,N] = A[M,K](f16) * BT[N,K](f16)^T + bias
//   mode 0: C f32 store (bias added)
//   mode 1: exact-erf GELU(C + bias) -> f16 store
//
// Block: 128 threads (4 wave32 waves, 2x2 wave grid), 128x128 block tile.
// Per wave: 64x64 output = 4x4 tiles of 16x16 via v_wmma_f32_16x16x32_f16.
// A/B tiles (128x32 f16 each) double-buffered in LDS, filled with
// GLOBAL_LOAD_ASYNC_TO_LDS_B128 (ASYNCcnt) when available.
// LDS rows padded to 40 halves (80B) to spread b128 fragment reads.
// Requires M%128==0, N%128==0, K%32==0.
// ---------------------------------------------------------------------------
#define LDS_STRIDE 40   // f16 elements per padded 32-element row

__global__ __launch_bounds__(128, 1)
void gemm_wmma_f16_kernel(const _Float16* __restrict__ A,
                          const _Float16* __restrict__ BT,
                          const float* __restrict__ bias,
                          float* __restrict__ Cf,
                          _Float16* __restrict__ Ch,
                          int Mdim, int Ndim, int Kdim, int mode) {
    __shared__ alignas(16) _Float16 Asm[2][128 * LDS_STRIDE];
    __shared__ alignas(16) _Float16 Bsm[2][128 * LDS_STRIDE];

    const int tid   = threadIdx.x;
    const int lane  = tid & 31;
    const int wid   = tid >> 5;       // 0..3
    const int half  = lane >> 4;      // 0/1 (lane group)
    const int r     = lane & 15;
    const int waveM = wid >> 1;       // 0..1
    const int waveN = wid & 1;        // 0..1
    const int blockM = blockIdx.y * 128;
    const int blockN = blockIdx.x * 128;
    const int rowBase = blockM + waveM * 64;   // + m*16 (+ half*8 + v)
    const int colBase = blockN + waveN * 64;   // + n*16 + r

    // Cooperative tile-load mapping: 512 chunks of 16B per tile, 4 per thread.
    // chunk c = i*128 + tid -> row = c>>2, col = (c&3)*8  (64B coalesced runs)
    const int ldRow0 = tid >> 2;          // advance by 32 per i
    const int ldCol  = (tid & 3) * 8;

    const v8f zero = {0.f, 0.f, 0.f, 0.f, 0.f, 0.f, 0.f, 0.f};
    v8f acc[4][4];
#pragma unroll
    for (int m = 0; m < 4; ++m)
#pragma unroll
        for (int n = 0; n < 4; ++n) acc[m][n] = zero;

    const int nk = Kdim >> 5;  // K/32 steps

    // Issue tile loads for K-step kt into buffer buf (8 copies per thread).
    auto issue_tile = [&](int buf, int k0) {
#pragma unroll
        for (int i = 0; i < 4; ++i) {
            const int row = ldRow0 + i * 32;
            copy16_g2lds(A + (size_t)(blockM + row) * Kdim + k0 + ldCol,
                         &Asm[buf][row * LDS_STRIDE + ldCol]);
        }
#pragma unroll
        for (int i = 0; i < 4; ++i) {
            const int row = ldRow0 + i * 32;
            copy16_g2lds(BT + (size_t)(blockN + row) * Kdim + k0 + ldCol,
                         &Bsm[buf][row * LDS_STRIDE + ldCol]);
        }
    };

    issue_tile(0, 0);

    for (int kt = 0; kt < nk; ++kt) {
        const int cur = kt & 1;
        if (kt + 1 < nk) {
            issue_tile(cur ^ 1, (kt + 1) << 5);
            WAIT_ASYNC(8);      // previous batch (cur) fully landed in LDS
        } else {
            WAIT_ASYNC(0);
        }
        __syncthreads();        // cur tile visible to all waves

        // B fragments: contiguous K order (same as A) — lanes 0-15 hold K=0..15,
        // lanes 16-31 hold K=16..31; column N = r (BT row is contiguous in K).
        F16x16 bfrag[4];
#pragma unroll
        for (int n = 0; n < 4; ++n) {
            const _Float16* p =
                &Bsm[cur][(waveN * 64 + n * 16 + r) * LDS_STRIDE + half * 16];
            bfrag[n].h8[0] = *(const v8h*)p;
            bfrag[n].h8[1] = *(const v8h*)(p + 8);
        }
        // A fragments: contiguous K order (same as B) —
        //   element i of lane: K = half*16 + i
#pragma unroll
        for (int m = 0; m < 4; ++m) {
            F16x16 afrag;
            const _Float16* p =
                &Asm[cur][(waveM * 64 + m * 16 + r) * LDS_STRIDE + half * 16];
            afrag.h8[0] = *(const v8h*)p;
            afrag.h8[1] = *(const v8h*)(p + 8);
#pragma unroll
            for (int n = 0; n < 4; ++n)
                acc[m][n] = __builtin_amdgcn_wmma_f32_16x16x32_f16(
                    false, afrag.v, false, bfrag[n].v, (short)0, acc[m][n],
                    false, false);
        }
        __syncthreads();        // all waves done reading cur before overwrite
    }

    // Epilogue. D layout: VGPR v -> row = base + v + half*8, col = base + r.
#pragma unroll
    for (int m = 0; m < 4; ++m) {
#pragma unroll
        for (int n = 0; n < 4; ++n) {
            const int col = colBase + n * 16 + r;
            const float bv = bias ? bias[col] : 0.0f;
#pragma unroll
            for (int v = 0; v < 8; ++v) {
                const int row = rowBase + m * 16 + half * 8 + v;
                const float val = acc[m][n][v] + bv;
                if (mode == 0) {
                    Cf[(size_t)row * Ndim + col] = val;
                } else {
                    // exact GELU: 0.5*x*(1+erf(x/sqrt(2)))
                    const float g =
                        0.5f * val * (1.0f + erff(val * 0.7071067811865475f));
                    Ch[(size_t)row * Ndim + col] = (_Float16)g;
                }
            }
        }
    }
}

// ---------------------------------------------------------------------------
// Pair attention: num_heads = S/2, tph = 2. One block per (batch, pair).
// Tokens i0 = perm[2p], i1 = perm[2p+1]; 2x2 scores over H dims, softmax,
// blend V. Writing results back to rows i0/i1 implements the inv_perm gather
// (inv_perm[perm[i]] == i). Output stored f16 (feeds the O-projection GEMM).
// ---------------------------------------------------------------------------
__global__ __launch_bounds__(256)
void attn_pair_kernel(const float* __restrict__ Q,
                      const float* __restrict__ Km,
                      const float* __restrict__ V,
                      const int* __restrict__ perm,
                      _Float16* __restrict__ att) {
    const int pairs = cfg::S / 2;
    const int b = blockIdx.x / pairs;
    const int p = blockIdx.x % pairs;
    const int i0 = perm[2 * p];
    const int i1 = perm[2 * p + 1];
    const size_t r0 = ((size_t)b * cfg::S + i0) * cfg::H;
    const size_t r1 = ((size_t)b * cfg::S + i1) * cfg::H;

    float s00 = 0.f, s01 = 0.f, s10 = 0.f, s11 = 0.f;
    for (int i = threadIdx.x; i < cfg::H; i += 256) {
        const float q0 = Q[r0 + i], q1 = Q[r1 + i];
        const float k0 = Km[r0 + i], k1 = Km[r1 + i];
        s00 += q0 * k0; s01 += q0 * k1;
        s10 += q1 * k0; s11 += q1 * k1;
    }

    __shared__ float red[4][256];
    __shared__ float coef[4];
    red[0][threadIdx.x] = s00; red[1][threadIdx.x] = s01;
    red[2][threadIdx.x] = s10; red[3][threadIdx.x] = s11;
    __syncthreads();
    for (int t = 128; t > 0; t >>= 1) {
        if (threadIdx.x < t) {
#pragma unroll
            for (int j = 0; j < 4; ++j)
                red[j][threadIdx.x] += red[j][threadIdx.x + t];
        }
        __syncthreads();
    }
    if (threadIdx.x == 0) {
        const float scale = 0.03125f;  // 1/sqrt(1024)
        const float a0 = red[0][0] * scale, a1 = red[1][0] * scale;
        const float b0 = red[2][0] * scale, b1 = red[3][0] * scale;
        const float m0 = fmaxf(a0, a1), m1 = fmaxf(b0, b1);
        const float e00 = expf(a0 - m0), e01 = expf(a1 - m0);
        const float e10 = expf(b0 - m1), e11 = expf(b1 - m1);
        const float d0 = 1.0f / (e00 + e01), d1 = 1.0f / (e10 + e11);
        coef[0] = e00 * d0; coef[1] = e01 * d0;
        coef[2] = e10 * d1; coef[3] = e11 * d1;
    }
    __syncthreads();
    const float a00 = coef[0], a01 = coef[1], a10 = coef[2], a11 = coef[3];
    for (int i = threadIdx.x; i < cfg::H; i += 256) {
        const float v0 = V[r0 + i], v1 = V[r1 + i];
        att[r0 + i] = (_Float16)(a00 * v0 + a01 * v1);
        att[r1 + i] = (_Float16)(a10 * v0 + a11 * v1);
    }
}

// ---------------------------------------------------------------------------
// Residual add + LayerNorm over H=1024. One block (256 thr) per row.
// Writes f32 result; optional f16 copy for the following GEMM.
// ---------------------------------------------------------------------------
__global__ __launch_bounds__(256)
void resid_ln_kernel(const float* __restrict__ xin,
                     const float* __restrict__ yin,
                     const float* __restrict__ g,
                     const float* __restrict__ be,
                     float* __restrict__ outf,
                     _Float16* __restrict__ outh) {
    const size_t row = blockIdx.x;
    const float* xr = xin + row * cfg::H;
    const float* yr = yin + row * cfg::H;

    float vals[4];
    float s = 0.f;
#pragma unroll
    for (int j = 0; j < 4; ++j) {
        const int i = threadIdx.x + j * 256;
        vals[j] = xr[i] + yr[i];
        s += vals[j];
    }
    __shared__ float red[256];
    red[threadIdx.x] = s;
    __syncthreads();
    for (int t = 128; t > 0; t >>= 1) {
        if (threadIdx.x < t) red[threadIdx.x] += red[threadIdx.x + t];
        __syncthreads();
    }
    const float mu = red[0] * (1.0f / cfg::H);
    __syncthreads();

    float s2 = 0.f;
#pragma unroll
    for (int j = 0; j < 4; ++j) {
        const float d = vals[j] - mu;
        s2 += d * d;
    }
    red[threadIdx.x] = s2;
    __syncthreads();
    for (int t = 128; t > 0; t >>= 1) {
        if (threadIdx.x < t) red[threadIdx.x] += red[threadIdx.x + t];
        __syncthreads();
    }
    const float rs = rsqrtf(red[0] * (1.0f / cfg::H) + cfg::LN_EPS);

#pragma unroll
    for (int j = 0; j < 4; ++j) {
        const int i = threadIdx.x + j * 256;
        const float o = (vals[j] - mu) * rs * g[i] + be[i];
        outf[row * cfg::H + i] = o;
        if (outh) outh[row * cfg::H + i] = (_Float16)o;
    }
}

// ---------------------------------------------------------------------------
// Host-side orchestration
// ---------------------------------------------------------------------------
extern "C" void kernel_launch(void* const* d_in, const int* in_sizes, int n_in,
                              void* d_out, int out_size, void* d_ws,
                              size_t ws_size, hipStream_t stream) {
    using namespace cfg;
    (void)in_sizes; (void)n_in; (void)out_size; (void)ws_size;

    const float* x   = (const float*)d_in[0];
    const float* Wq  = (const float*)d_in[1];
    const float* bq  = (const float*)d_in[2];
    const float* Wk  = (const float*)d_in[3];
    const float* bk  = (const float*)d_in[4];
    const float* Wv  = (const float*)d_in[5];
    const float* bv  = (const float*)d_in[6];
    const float* Wo  = (const float*)d_in[7];
    const float* bo  = (const float*)d_in[8];
    const float* g1  = (const float*)d_in[9];
    const float* be1 = (const float*)d_in[10];
    const float* g2  = (const float*)d_in[11];
    const float* be2 = (const float*)d_in[12];
    const float* W1  = (const float*)d_in[13];
    const float* b1  = (const float*)d_in[14];
    const float* W2  = (const float*)d_in[15];
    const float* b2  = (const float*)d_in[16];
    const int*   perm = (const int*)d_in[17];

    // Workspace layout (bytes); peak ~232 MB.
    char* ws = (char*)d_ws;
    _Float16* xh  = (_Float16*)(ws + ((size_t)0 << 20));    // 16 MB
    _Float16* WqT = (_Float16*)(ws + ((size_t)16 << 20));   //  2 MB
    _Float16* WkT = (_Float16*)(ws + ((size_t)18 << 20));   //  2 MB
    _Float16* WvT = (_Float16*)(ws + ((size_t)20 << 20));   //  2 MB
    _Float16* WoT = (_Float16*)(ws + ((size_t)22 << 20));   //  2 MB
    _Float16* W1T = (_Float16*)(ws + ((size_t)24 << 20));   //  8 MB
    _Float16* W2T = (_Float16*)(ws + ((size_t)32 << 20));   //  8 MB
    float*    Qb  = (float*)(ws + ((size_t)40 << 20));      // 32 MB
    float*    Kb  = (float*)(ws + ((size_t)72 << 20));      // 32 MB
    float*    Vb  = (float*)(ws + ((size_t)104 << 20));     // 32 MB
    _Float16* att = (_Float16*)(ws + ((size_t)136 << 20));  // 16 MB
    float*    ybuf = (float*)(ws + ((size_t)152 << 20));    // 32 MB
    float*    x1  = (float*)(ws + ((size_t)184 << 20));     // 32 MB
    _Float16* x1h = (_Float16*)(ws + ((size_t)216 << 20));  // 16 MB
    // FFN intermediate (64 MB) reuses the dead Q/K region.
    _Float16* hf  = (_Float16*)(ws + ((size_t)40 << 20));

    const dim3 tblk(32, 8);

    // 1. Convert x to f16.
    cvt_f32_to_f16_kernel<<<4096, 256, 0, stream>>>(x, xh, (size_t)M * H);

    // 2. Transpose + convert weights: W[K][N] -> WT[N][K] f16.
    transpose_cvt_kernel<<<dim3(H / 32, H / 32), tblk, 0, stream>>>(Wq, WqT, H, H);
    transpose_cvt_kernel<<<dim3(H / 32, H / 32), tblk, 0, stream>>>(Wk, WkT, H, H);
    transpose_cvt_kernel<<<dim3(H / 32, H / 32), tblk, 0, stream>>>(Wv, WvT, H, H);
    transpose_cvt_kernel<<<dim3(H / 32, H / 32), tblk, 0, stream>>>(Wo, WoT, H, H);
    transpose_cvt_kernel<<<dim3(F / 32, H / 32), tblk, 0, stream>>>(W1, W1T, H, F);
    transpose_cvt_kernel<<<dim3(H / 32, F / 32), tblk, 0, stream>>>(W2, W2T, F, H);

    // 3. Q/K/V projections (f32 out + bias).
    const dim3 gHH(H / 128, M / 128);
    gemm_wmma_f16_kernel<<<gHH, 128, 0, stream>>>(xh, WqT, bq, Qb, nullptr, M, H, H, 0);
    gemm_wmma_f16_kernel<<<gHH, 128, 0, stream>>>(xh, WkT, bk, Kb, nullptr, M, H, H, 0);
    gemm_wmma_f16_kernel<<<gHH, 128, 0, stream>>>(xh, WvT, bv, Vb, nullptr, M, H, H, 0);

    // 4. Pair attention (incl. permutation + inverse permutation), f16 out.
    attn_pair_kernel<<<Bn * (S / 2), 256, 0, stream>>>(Qb, Kb, Vb, perm, att);

    // 5. Output projection.
    gemm_wmma_f16_kernel<<<gHH, 128, 0, stream>>>(att, WoT, bo, ybuf, nullptr, M, H, H, 0);

    // 6. x1 = LN(x + attended); f32 + f16 copies.
    resid_ln_kernel<<<M, 256, 0, stream>>>(x, ybuf, g1, be1, x1, x1h);

    // 7. FFN up-projection with exact GELU, f16 out.
    gemm_wmma_f16_kernel<<<dim3(F / 128, M / 128), 128, 0, stream>>>(
        x1h, W1T, b1, nullptr, hf, M, F, H, 1);

    // 8. FFN down-projection, f32 out.
    gemm_wmma_f16_kernel<<<gHH, 128, 0, stream>>>(hf, W2T, b2, ybuf, nullptr, M, H, F, 0);

    // 9. out = LN(x1 + ffn).
    resid_ln_kernel<<<M, 256, 0, stream>>>(x1, ybuf, g2, be2, (float*)d_out, nullptr);
}
